// miScore_6030134083663
// MI455X (gfx1250) — matrix operand order repaired and checked by reading, not compiled, not run on hardware
//
#include <hip/hip_runtime.h>

typedef __attribute__((ext_vector_type(16))) _Float16 v16h;
typedef __attribute__((ext_vector_type(8)))  _Float16 v8h;
typedef __attribute__((ext_vector_type(8)))  float    v8f;

#define NUM_USR 8192
#define HDIM    256
#define JT_TOTAL (NUM_USR / 16)        // 512 j-tiles
#define JSLICES  16                    // j split for occupancy
#define JT_PER   (JT_TOTAL / JSLICES)  // 32 j-tiles per block
#define ROWS_PER_BLOCK 256             // 8 waves x 2 i-tiles x 16 rows
#define IBLOCKS  (NUM_USR / ROWS_PER_BLOCK)  // 32
#define LDS_ROW  264                   // 256 halves + 8 pad (16B) -> bank-conflict-free

// ---- CDNA5 async copy to LDS (ASYNCcnt path) ------------------------------
__device__ __forceinline__ void async_copy_b128(unsigned lds_off, const _Float16* g) {
  asm volatile("global_load_async_to_lds_b128 %0, %1, off"
               :: "v"(lds_off), "v"(g) : "memory");
}
__device__ __forceinline__ void wait_async0() {
  asm volatile("s_wait_asynccnt 0x0" ::: "memory");
}

// ---------------------------------------------------------------------------
// Kernel 1: proj_h[j,s] = (f16)( sum_t y_b[j,t] * W[s,t] + b[s] )
// ---------------------------------------------------------------------------
__global__ __launch_bounds__(256)
void proj_kernel(const float* __restrict__ y_b, const float* __restrict__ W,
                 const float* __restrict__ bias, _Float16* __restrict__ proj_h) {
  const int lane    = threadIdx.x & 31;
  const int wave    = threadIdx.x >> 5;
  const int tile    = blockIdx.x * 8 + wave;
  const int jbase   = (tile >> 4) * 16;
  const int sbase   = (tile & 15) * 16;
  const int halfSel = lane >> 4;
  const int lid     = lane & 15;

  const float* brow = y_b + (size_t)(jbase + lid) * HDIM;
  const float* wrow = W   + (size_t)(sbase + lid) * HDIM;

  v8f c = {};
  for (int k = 0; k < HDIM; k += 32) {
    v16h a, bm;
    const int kbA = k + halfSel * 8;
    const int kbB = k + halfSel * 16;
#pragma unroll
    for (int i = 0; i < 8; ++i) {
      a[i]     = (_Float16)brow[kbA + i];
      a[i + 8] = (_Float16)brow[kbA + 16 + i];
    }
#pragma unroll
    for (int i = 0; i < 16; ++i) bm[i] = (_Float16)wrow[kbA + (i & 7) + (i >> 3) * 16];
    c = __builtin_amdgcn_wmma_f32_16x16x32_f16(false, a, false, bm,
                                               (short)0, c, false, false);
  }

  const float bv = bias[sbase + lid];
#pragma unroll
  for (int r = 0; r < 8; ++r) {
    const int j = jbase + r + halfSel * 8;
    proj_h[(size_t)j * HDIM + sbase + lid] = (_Float16)(c[r] + bv);
  }
}

// ---------------------------------------------------------------------------
// Kernel 2: block = 256 rows (8 waves x 2 i-tiles), B tile double-buffered in
// LDS via async-to-LDS; each 8KB B tile feeds 128 WMMAs. Partial row sums per
// j-slice written with exclusive ownership (deterministic, no atomics).
// ---------------------------------------------------------------------------
__global__ __launch_bounds__(256)
void score_kernel(const float* __restrict__ y_a,
                  const _Float16* __restrict__ proj_h,
                  float* __restrict__ partial) {
  __shared__ _Float16 tile[2][16][LDS_ROW];

  const int tid     = threadIdx.x;
  const int lane    = tid & 31;
  const int wave    = tid >> 5;
  const int halfSel = lane >> 4;
  const int lid     = lane & 15;

  const int ibase = blockIdx.x * ROWS_PER_BLOCK;
  const int jt0   = blockIdx.y * JT_PER;

  // A fragments for this wave's two 16-row tiles (pinned in registers)
  v16h afrag0[8], afrag1[8];
  {
    const float* ar0 = y_a + (size_t)(ibase + wave * 16 + lid) * HDIM;
    const float* ar1 = ar0 + 128 * HDIM;
#pragma unroll
    for (int kk = 0; kk < 8; ++kk) {
      const int kb = kk * 32 + halfSel * 8;
#pragma unroll
      for (int i = 0; i < 8; ++i) {
        afrag0[kk][i]     = (_Float16)ar0[kb + i];
        afrag0[kk][i + 8] = (_Float16)ar0[kb + 16 + i];
        afrag1[kk][i]     = (_Float16)ar1[kb + i];
        afrag1[kk][i + 8] = (_Float16)ar1[kb + 16 + i];
      }
    }
  }

  // cooperative async prefetch: 256 threads x 32B cover the 8KB tile
  const int prow = tid >> 4;         // 0..15 (j row within tile)
  const int pcol = (tid & 15) * 16;  // halves
  auto prefetch = [&](int buf, int jt) {
    const _Float16* g = proj_h + (size_t)(jt * 16 + prow) * HDIM + pcol;
    unsigned l = (unsigned)(size_t)&tile[buf][prow][pcol];
    async_copy_b128(l,      g);
    async_copy_b128(l + 16, g + 8);
  };

  prefetch(0, jt0);

  v8f sums0 = {}, sums1 = {};
  for (int t = 0; t < JT_PER; ++t) {
    wait_async0();
    __syncthreads();
    if (t + 1 < JT_PER) prefetch((t + 1) & 1, jt0 + t + 1);

    const _Float16* bp = &tile[t & 1][lid][halfSel * 8];
    v8f c0 = {}, c1 = {};
#pragma unroll
    for (int kk = 0; kk < 8; ++kk) {
      v8h lo = *(const v8h*)(bp + kk * 32);
      v8h hi = *(const v8h*)(bp + kk * 32 + 16);
      v16h bf = __builtin_shufflevector(lo, hi, 0, 1, 2, 3, 4, 5, 6, 7,
                                        8, 9, 10, 11, 12, 13, 14, 15);
      c0 = __builtin_amdgcn_wmma_f32_16x16x32_f16(false, afrag0[kk], false, bf,
                                                  (short)0, c0, false, false);
      c1 = __builtin_amdgcn_wmma_f32_16x16x32_f16(false, afrag1[kk], false, bf,
                                                  (short)0, c1, false, false);
    }
#pragma unroll
    for (int r = 0; r < 8; ++r) {
      sums0[r] += 1.0f / (1.0f + __expf(-c0[r]));
      sums1[r] += 1.0f / (1.0f + __expf(-c1[r]));
    }
  }

  // reduce over the 16 columns (intra-half shuffles)
#pragma unroll
  for (int m = 1; m < 16; m <<= 1) {
#pragma unroll
    for (int r = 0; r < 8; ++r) {
      sums0[r] += __shfl_xor(sums0[r], m, 32);
      sums1[r] += __shfl_xor(sums1[r], m, 32);
    }
  }
  if (lid == 0) {
    float* dst = partial + (size_t)blockIdx.y * NUM_USR;
    const int r0 = ibase + wave * 16 + halfSel * 8;
#pragma unroll
    for (int r = 0; r < 8; ++r) {
      dst[r0 + r]       = sums0[r];
      dst[r0 + 128 + r] = sums1[r];
    }
  }
}

// ---------------------------------------------------------------------------
// Kernel 3: out[i, :] = (1/256) * sum_slices partial[s][i]
// ---------------------------------------------------------------------------
__global__ __launch_bounds__(256)
void finalize_kernel(const float* __restrict__ partial, float* __restrict__ out) {
  __shared__ float sv[16];
  const int ibase = blockIdx.x * 16;
  if (threadIdx.x < 16) {
    float s = 0.0f;
#pragma unroll
    for (int k = 0; k < JSLICES; ++k)
      s += partial[(size_t)k * NUM_USR + ibase + threadIdx.x];
    sv[threadIdx.x] = s * (1.0f / 256.0f);
  }
  __syncthreads();
  const int row = threadIdx.x >> 4;
  const float v = sv[row];
  float4 v4 = make_float4(v, v, v, v);
  float4* op = (float4*)(out + (size_t)(ibase + row) * HDIM +
                         (threadIdx.x & 15) * 16);
#pragma unroll
  for (int q = 0; q < 4; ++q) op[q] = v4;
}

// ---------------------------------------------------------------------------
extern "C" void kernel_launch(void* const* d_in, const int* in_sizes, int n_in,
                              void* d_out, int out_size, void* d_ws, size_t ws_size,
                              hipStream_t stream) {
  const float* y_a = (const float*)d_in[0];
  const float* y_b = (const float*)d_in[1];
  const float* W   = (const float*)d_in[2];
  const float* b   = (const float*)d_in[3];
  float*       out = (float*)d_out;

  _Float16* proj_h = (_Float16*)d_ws;                       // 4 MB
  float* partial = (float*)((char*)d_ws +
                            (size_t)NUM_USR * HDIM * sizeof(_Float16));  // 512 KB

  proj_kernel<<<(NUM_USR / 16) * (HDIM / 16) / 8, 256, 0, stream>>>(y_b, W, b, proj_h);

  dim3 grid(IBLOCKS, JSLICES);
  score_kernel<<<grid, 256, 0, stream>>>(y_a, proj_h, partial);

  finalize_kernel<<<NUM_USR / 16, 256, 0, stream>>>(partial, out);
}
